// PairEmbed_4569845203028
// MI455X (gfx1250) — matrix operand order repaired and checked by reading, not compiled, not run on hardware
//
#include <hip/hip_runtime.h>

typedef _Float16 half_t;
typedef __attribute__((ext_vector_type(8)))  _Float16 v8h;
typedef __attribute__((ext_vector_type(16))) _Float16 v16h;
typedef __attribute__((ext_vector_type(2)))  float    v2f;
typedef __attribute__((ext_vector_type(4)))  float    v4f;
typedef __attribute__((ext_vector_type(8)))  float    v8f;

#define BATCH   128
#define LSEQ    128
#define NPAIR   8256   // 128*129/2
#define EPSF    1e-8f
#define BNEPS   1e-5f
#define PI_F     3.14159265358979323846f
#define TWO_PI_F 6.28318530717958647692f
#define COLS_PER_ITER 128
#define ITERS_PER_BLOCK 4
#define COLS_PER_BLOCK (COLS_PER_ITER * ITERS_PER_BLOCK)   // 512

__device__ __forceinline__ v16h cat16(v8h a, v8h b) {
  return __builtin_shufflevector(a, b, 0,1,2,3,4,5,6,7,8,9,10,11,12,13,14,15);
}

__device__ __forceinline__ float bninv(float g, float v) { return g / sqrtf(v + BNEPS); }

// Branch-free GELU: v * sigmoid(1.5957691*v + 0.0713548*v^3)  (tanh-form), err <~3e-4,
// below the f16 activation quantization of the WMMA path. 2 TRANS ops, no divergence.
__device__ __forceinline__ float gelu_fast(float v) {
  float q = v * (1.5957691216f + 0.0713548163f * v * v);
  float r = __builtin_amdgcn_rcpf(1.0f + __expf(-q));
  return v * r;
}

// Branchless atan2, ~1e-7 rad (minimax atan poly on [0,1] + quadrant selects).
__device__ __forceinline__ float fast_atan2f(float y, float x) {
  float ax = __builtin_fabsf(x), ay = __builtin_fabsf(y);
  float mx = fmaxf(ax, ay), mn = fminf(ax, ay);
  float a  = mn * __builtin_amdgcn_rcpf(mx);
  float s  = a * a;
  float r  =               2.78569828e-3f;
  r = fmaf(r, s, -1.58660226e-2f);
  r = fmaf(r, s,  4.24722321e-2f);
  r = fmaf(r, s, -7.49753043e-2f);
  r = fmaf(r, s,  1.06448799e-1f);
  r = fmaf(r, s, -1.42070308e-1f);
  r = fmaf(r, s,  1.99934542e-1f);
  r = fmaf(r, s, -3.33331466e-1f);
  r = fmaf(r * s, a, a);
  float t = (ay > ax) ? (1.57079637f - r) : r;
  t = (x < 0.0f) ? (3.14159274f - t) : t;
  return copysignf(t, y);
}

// Layer 0: 64x4 x (4 x 16 cols) via fp32 WMMA 16x16x4, + bias + GELU, f16 out.
__device__ __forceinline__ void layer0_wmma(const float* F, const float* W0f,
                                            const float* bias, half_t* outAct,
                                            int ncol0, int lane)
{
  const int m16 = lane & 15;
  const int hi  = lane >> 4;
  const int n   = ncol0 + m16;
  // B 4x16 f32: VGPR0 = K (hi*2), VGPR1 = K (hi*2+1), N = lane%16
  v2f bfrag = *(const v2f*)(F + n*4 + hi*2);
  #pragma unroll
  for (int mt = 0; mt < 4; ++mt) {
    // A 16x4 f32: VGPR0 = K (hi*2), VGPR1 = K (hi*2+1), M row = mt*16 + lane%16
    v2f afrag = *(const v2f*)(W0f + (mt*16 + m16)*4 + hi*2);
    v8f c = {0.f,0.f,0.f,0.f,0.f,0.f,0.f,0.f};
    c = __builtin_amdgcn_wmma_f32_16x16x4_f32(false, afrag, false, bfrag, (short)0, c, false, false);
    const int mbase = mt*16 + hi*8;
    v8h o;
    #pragma unroll
    for (int r = 0; r < 8; ++r) {
      float v = gelu_fast(c[r] + bias[mbase + r]);
      o[r] = (half_t)v;
    }
    *(v8h*)(outAct + n*72 + mbase) = o;
  }
}

// One 64-out linear layer on a 16-column group, via WMMA 16x16x32 f16.
template<bool GELU, int MT>
__device__ __forceinline__ void layer_wmma(const half_t* act, const half_t* W,
                                           const float* bias, half_t* outAct,
                                           int ncol0, int lane)
{
  const int m16 = lane & 15;
  const int hi  = lane >> 4;
  const int n   = ncol0 + m16;
  const half_t* bp = act + n*72 + hi*8;
  v16h bf0 = cat16(*(const v8h*)bp,        *(const v8h*)(bp + 16));
  v16h bf1 = cat16(*(const v8h*)(bp + 32), *(const v8h*)(bp + 48));
  #pragma unroll
  for (int mt = 0; mt < MT; ++mt) {
    const half_t* ap = W + (mt*16 + m16)*72 + hi*8;
    v16h a0 = cat16(*(const v8h*)ap,        *(const v8h*)(ap + 16));
    v16h a1 = cat16(*(const v8h*)(ap + 32), *(const v8h*)(ap + 48));
    v8f c = {0.f,0.f,0.f,0.f,0.f,0.f,0.f,0.f};
    c = __builtin_amdgcn_wmma_f32_16x16x32_f16(false, a0, false, bf0, (short)0, c, false, false);
    c = __builtin_amdgcn_wmma_f32_16x16x32_f16(false, a1, false, bf1, (short)0, c, false, false);
    const int mbase = mt*16 + hi*8;
    v8h o;
    #pragma unroll
    for (int r = 0; r < 8; ++r) {
      float v = c[r] + bias[mbase + r];
      if (GELU) v = gelu_fast(v);
      o[r] = (half_t)v;
    }
    *(v8h*)(outAct + n*72 + mbase) = o;
  }
}

__device__ __forceinline__ void pair_from_index(int p, int& i, int& j) {
  i = (int)((__builtin_amdgcn_sqrtf(8.0f*(float)p + 1.0f) - 1.0f) * 0.5f);
  while ((i+1)*(i+2)/2 <= p) ++i;
  while (i*(i+1)/2 > p) --i;
  j = p - i*(i+1)/2;
}

__global__ __launch_bounds__(256) void pairembed_kernel(
    const float* __restrict__ x,
    const float* __restrict__ W0, const float* __restrict__ b0,
    const float* __restrict__ W1, const float* __restrict__ b1,
    const float* __restrict__ W2, const float* __restrict__ b2,
    const float* __restrict__ W3, const float* __restrict__ b3,
    const float* __restrict__ g0, const float* __restrict__ be0, const float* __restrict__ mu0, const float* __restrict__ va0,
    const float* __restrict__ g1, const float* __restrict__ be1, const float* __restrict__ mu1, const float* __restrict__ va1,
    const float* __restrict__ g2, const float* __restrict__ be2, const float* __restrict__ mu2, const float* __restrict__ va2,
    const float* __restrict__ g3, const float* __restrict__ be3, const float* __restrict__ mu3, const float* __restrict__ va3,
    const float* __restrict__ g4, const float* __restrict__ be4, const float* __restrict__ mu4, const float* __restrict__ va4,
    float* __restrict__ y)
{
  __shared__ __align__(16) half_t sAct0[128*72];  // layer0 out / layer2 out
  __shared__ __align__(16) half_t sAct1[128*72];  // layer1 out
  __shared__ __align__(16) half_t sW1[64*72];
  __shared__ __align__(16) half_t sW2[64*72];
  __shared__ __align__(16) half_t sW3[16*72];     // padded 8->16 rows
  __shared__ __align__(16) float  sW0[256];       // BN0+BN1 folded, f32 [64][4]
  __shared__ __align__(16) float  sF[128*4];      // per-column features, f32 [col][4]
  __shared__ float sb0[64], sb1[64], sb2[64], sb3[16];

  const int tid = threadIdx.x;
  const int b   = blockIdx.y;

  // ---- Fold all BatchNorms into weights/biases, stage f16 weights in LDS (once) ----
  for (int idx = tid; idx < 4096; idx += 256) {
    int m = idx >> 6, k = idx & 63;
    sW1[m*72 + k] = (half_t)(bninv(g2[m], va2[m]) * W1[idx]);
    sW2[m*72 + k] = (half_t)(bninv(g3[m], va3[m]) * W2[idx]);
  }
  for (int idx = tid; idx < 1024; idx += 256) {
    int m = idx >> 6, k = idx & 63;
    sW3[m*72 + k] = (m < 8) ? (half_t)(bninv(g4[m], va4[m]) * W3[m*64 + k]) : (half_t)0.0f;
  }
  {
    int m = tid >> 2, c = tid & 3;
    sW0[tid] = bninv(g1[m], va1[m]) * W0[tid] * bninv(g0[c], va0[c]);
  }
  if (tid < 64) {
    int m = tid;
    float inv1 = bninv(g1[m], va1[m]);
    float kacc = 0.0f;
    #pragma unroll
    for (int c = 0; c < 4; ++c) {
      float inv0 = bninv(g0[c], va0[c]);
      kacc += W0[m*4 + c] * (be0[c] - mu0[c]*inv0);
    }
    sb0[m] = inv1 * (b0[m] + kacc - mu1[m]) + be1[m];
    sb1[m] = bninv(g2[m], va2[m]) * (b1[m] - mu2[m]) + be2[m];
    sb2[m] = bninv(g3[m], va3[m]) * (b2[m] - mu3[m]) + be3[m];
  }
  if (tid < 16)
    sb3[tid] = (tid < 8) ? (bninv(g4[tid], va4[tid]) * (b3[tid] - mu4[tid]) + be4[tid]) : 0.0f;
  __syncthreads();

  const int lane  = tid & 31;
  const int ncol0 = (tid >> 5) * 16;   // 8 waves x 16 columns
  const float* xb = x + (size_t)b * 512;
  float* yb = y + (size_t)b * 8 * 16384;

  for (int it = 0; it < ITERS_PER_BLOCK; ++it) {
    const int p0 = blockIdx.x * COLS_PER_BLOCK + it * COLS_PER_ITER;
    if (p0 >= NPAIR) break;

    // ---- Stage 1: pairwise features (branchless, native transcendentals) ----
    if (tid < COLS_PER_ITER) {
      const int col = tid;
      int p = p0 + col; if (p > NPAIR-1) p = NPAIR-1;
      int i, j; pair_from_index(p, i, j);
      float pxi = xb[i],       pyi = xb[128+i], pzi = xb[256+i], ei = xb[384+i];
      float pxj = xb[j],       pyj = xb[128+j], pzj = xb[256+j], ej = xb[384+j];
      float pti = __builtin_amdgcn_sqrtf(pxi*pxi + pyi*pyi);
      float ptj = __builtin_amdgcn_sqrtf(pxj*pxj + pyj*pyj);
      // rap = 0.5*log1p(2pz/max(e-pz,eps)) = 0.5*log((e+pz)/max(e-pz,eps))
      float rapi = 0.5f * __logf((ei + pzi) * __builtin_amdgcn_rcpf(fmaxf(ei - pzi, 1e-20f)));
      float rapj = 0.5f * __logf((ej + pzj) * __builtin_amdgcn_rcpf(fmaxf(ej - pzj, 1e-20f)));
      float phii = fast_atan2f(pyi, pxi);
      float phij = fast_atan2f(pyj, pxj);
      // a = phii - phij + pi  is in (-pi, 3pi): explicit mod-2pi (divisor-sign semantics)
      float a = phii - phij + PI_F;
      if (a < 0.0f)       a += TWO_PI_F;
      if (a >= TWO_PI_F)  a -= TWO_PI_F;
      float dphi = a - PI_F;
      float drap  = rapi - rapj;
      float delta = __builtin_amdgcn_sqrtf(drap*drap + dphi*dphi);
      float ptmin = fminf(pti, ptj);
      v4f f;
      f.x = __logf(fmaxf(ptmin * delta, EPSF));                                        // lnkt
      f.y = __logf(fmaxf(ptmin * __builtin_amdgcn_rcpf(fmaxf(pti + ptj, EPSF)), EPSF)); // lnz
      f.z = __logf(fmaxf(delta, EPSF));                                                // lndelta
      float sx = pxi+pxj, sy = pyi+pyj, sz = pzi+pzj, se = ei+ej;
      f.w = __logf(fmaxf(se*se - sx*sx - sy*sy - sz*sz, EPSF));                        // lnm2
      *(v4f*)(sF + col*4) = f;
    }
    __syncthreads();

    // ---- Layer 0: fp32 WMMA 16x16x4 + GELU ----
    layer0_wmma(sF, sW0, sb0, sAct0, ncol0, lane);
    __syncthreads();

    // ---- Layers 1 and 2: 64x64 GEMM via f16 WMMA + fast GELU ----
    layer_wmma<true, 4>(sAct0, sW1, sb1, sAct1, ncol0, lane);
    __syncthreads();
    layer_wmma<true, 4>(sAct1, sW2, sb2, sAct0, ncol0, lane);
    __syncthreads();

    // ---- Layer 3: (8->16 padded) x 64, no GELU, symmetric global scatter ----
    {
      const int m16 = lane & 15;
      const int hi  = lane >> 4;
      const int n   = ncol0 + m16;
      const half_t* bp = sAct0 + n*72 + hi*8;
      v16h bf0 = cat16(*(const v8h*)bp,        *(const v8h*)(bp + 16));
      v16h bf1 = cat16(*(const v8h*)(bp + 32), *(const v8h*)(bp + 48));
      const half_t* ap = sW3 + m16*72 + hi*8;
      v16h a0 = cat16(*(const v8h*)ap,        *(const v8h*)(ap + 16));
      v16h a1 = cat16(*(const v8h*)(ap + 32), *(const v8h*)(ap + 48));
      v8f c = {0.f,0.f,0.f,0.f,0.f,0.f,0.f,0.f};
      c = __builtin_amdgcn_wmma_f32_16x16x32_f16(false, a0, false, bf0, (short)0, c, false, false);
      c = __builtin_amdgcn_wmma_f32_16x16x32_f16(false, a1, false, bf1, (short)0, c, false, false);
      // lanes 0..15 hold M = 0..7 (all real channels), N = n
      if (hi == 0) {
        int p = p0 + n;
        if (p < NPAIR) {
          int i, j; pair_from_index(p, i, j);
          #pragma unroll
          for (int r = 0; r < 8; ++r) {
            float v = c[r] + sb3[r];
            yb[r*16384 + i*128 + j] = v;
            yb[r*16384 + j*128 + i] = v;
          }
        }
      }
    }
    __syncthreads();
  }
}

extern "C" void kernel_launch(void* const* d_in, const int* in_sizes, int n_in,
                              void* d_out, int out_size, void* d_ws, size_t ws_size,
                              hipStream_t stream) {
  (void)in_sizes; (void)n_in; (void)d_ws; (void)ws_size; (void)out_size;
  // setup_inputs() order: x, bn0{g,b,m,v}, W0,b0, bn1{..}, W1,b1, bn2{..}, W2,b2, bn3{..}, W3,b3, bn4{..}
  const float* x   = (const float*)d_in[0];
  const float* g0  = (const float*)d_in[1];
  const float* be0 = (const float*)d_in[2];
  const float* mu0 = (const float*)d_in[3];
  const float* va0 = (const float*)d_in[4];
  const float* W0  = (const float*)d_in[5];
  const float* b0  = (const float*)d_in[6];
  const float* g1  = (const float*)d_in[7];
  const float* be1 = (const float*)d_in[8];
  const float* mu1 = (const float*)d_in[9];
  const float* va1 = (const float*)d_in[10];
  const float* W1  = (const float*)d_in[11];
  const float* b1  = (const float*)d_in[12];
  const float* g2  = (const float*)d_in[13];
  const float* be2 = (const float*)d_in[14];
  const float* mu2 = (const float*)d_in[15];
  const float* va2 = (const float*)d_in[16];
  const float* W2  = (const float*)d_in[17];
  const float* b2  = (const float*)d_in[18];
  const float* g3  = (const float*)d_in[19];
  const float* be3 = (const float*)d_in[20];
  const float* mu3 = (const float*)d_in[21];
  const float* va3 = (const float*)d_in[22];
  const float* W3  = (const float*)d_in[23];
  const float* b3  = (const float*)d_in[24];
  const float* g4  = (const float*)d_in[25];
  const float* be4 = (const float*)d_in[26];
  const float* mu4 = (const float*)d_in[27];
  const float* va4 = (const float*)d_in[28];

  dim3 grid((NPAIR + COLS_PER_BLOCK - 1) / COLS_PER_BLOCK, BATCH);  // (17, 128)
  pairembed_kernel<<<grid, 256, 0, stream>>>(
      x, W0, b0, W1, b1, W2, b2, W3, b3,
      g0, be0, mu0, va0, g1, be1, mu1, va1,
      g2, be2, mu2, va2, g3, be3, mu3, va3,
      g4, be4, mu4, va4,
      (float*)d_out);
}
